// hbiascorrect_45509473468428
// MI455X (gfx1250) — matrix operand order repaired and checked by reading, not compiled, not run on hardware
//
#include <hip/hip_runtime.h>

// ---------------------------------------------------------------------------
// MI455X (gfx1250) implementation.
// Dense GEMMs: bf16 WMMA (v_wmma_f32_16x16x32_bf16, f32 accumulate) with
// Tensor-Data-Mover (tensor_load_to_lds) double-buffered LDS staging.
// LayerNorm / softmax / Sinkhorn / GRU recurrence / tiny g-encoder attention
// are f32 VALU kernels.  Workspace requirement: ~230 MB.
// ---------------------------------------------------------------------------

typedef __bf16 bf16_t;
typedef __attribute__((ext_vector_type(16))) __bf16 v16bf;
typedef __attribute__((ext_vector_type(8)))  float  v8f;
typedef __attribute__((ext_vector_type(4)))  unsigned u32x4_t;
typedef __attribute__((ext_vector_type(8)))  int      i32x8_t;
typedef __attribute__((ext_vector_type(4)))  int      i32x4_t;

static __device__ __forceinline__ bf16_t f2b(float f) {
  unsigned u = __builtin_bit_cast(unsigned, f);
  u += 0x7FFFu + ((u >> 16) & 1u);              // round-to-nearest-even
  return __builtin_bit_cast(bf16_t, (unsigned short)(u >> 16));
}
static __device__ __forceinline__ float sigm(float x) {
  return 1.0f / (1.0f + __expf(-x));
}

// ---------------- TDM: DMA one 64-row x 32-col bf16 tile into LDS ----------
// D# per CDNA5 ISA.  tensor_dim0 = k_rem and
// tensor_dim1 = rows_rem give hardware zero-fill for K tails and partial
// M/N tiles.
// This toolchain's builtin takes 6 args (g0 u32x4, g1 i32x8, g2 i32x4,
// g3 i32x4, extra i32x8, cpol i32).
static __device__ __forceinline__ void tdm_tile_load(
    unsigned lds_addr, const bf16_t* gptr, int rows_rem, int k_rem, long ld)
{
  const unsigned long ga = (unsigned long)(__SIZE_TYPE__)gptr;
  u32x4_t g0;
  g0.x = 1u;                                            // count=1 (valid)
  g0.y = lds_addr;                                      // LDS byte address
  g0.z = (unsigned)(ga & 0xffffffffu);                  // global_addr[31:0]
  g0.w = (unsigned)((ga >> 32) & 0x01ffffffu) | (2u << 30);  // [56:32]|type=2

  const unsigned td0 = (unsigned)k_rem;                 // tensor_dim0
  const unsigned td1 = (unsigned)rows_rem;              // tensor_dim1
  i32x8_t g1;
  g1[0] = (int)(1u << 16);                              // data_size=1 (2 bytes)
  g1[1] = (int)((td0 & 0xffffu) << 16);                 // abar=0 | td0.lo
  g1[2] = (int)((td0 >> 16) | ((td1 & 0xffffu) << 16)); // td0.hi | td1.lo
  g1[3] = (int)((td1 >> 16) | (32u << 16));             // td1.hi | tile_dim0=32
  g1[4] = 64;                                           // tile_dim1=64, dim2=0
  g1[5] = (int)(unsigned)ld;                            // dim0 stride (elems)
  g1[6] = 0;
  g1[7] = 0;

  const i32x4_t z4 = {0, 0, 0, 0};                      // 2-D tile: grp2/3 unused
  const i32x8_t z8 = {0, 0, 0, 0, 0, 0, 0, 0};
  __builtin_amdgcn_tensor_load_to_lds(g0, g1, z4, z4, z8, 0);
}

// --------------------------- f32 -> bf16 convert ---------------------------
__global__ void k_cvt(const float* __restrict__ x, bf16_t* __restrict__ y, long n) {
  long i  = (long)blockIdx.x * blockDim.x + threadIdx.x;
  long st = (long)gridDim.x * blockDim.x;
  for (; i < n; i += st) y[i] = f2b(x[i]);
}

// --------------------- batched transpose f32 -> bf16 -----------------------
// in: [Bt, R, C] f32   out: [Bt, C, R] bf16
__global__ void k_transpose_bf(const float* __restrict__ x, bf16_t* __restrict__ y,
                               int R, int C, long n) {
  long i  = (long)blockIdx.x * blockDim.x + threadIdx.x;
  long st = (long)gridDim.x * blockDim.x;
  for (; i < n; i += st) {
    long c = i % C;
    long t = i / C;
    long r = t % R;
    long bb = t / R;
    y[(bb * C + c) * R + r] = f2b(x[i]);
  }
}

// ------------------------------ WMMA GEMM ----------------------------------
// C[M,N] = act( A[M,K] @ W[N,K]^T + bias [+ Res] ), batched via blockIdx.z.
// A, W bf16; outputs optionally f32 (Cf) and/or bf16 (Cb), sharing ldc/sC.
// act: 0 = none, 1 = relu, 2 = sigmoid.
//
// Tiles are staged by the Tensor Data Mover (wave 0 issues two
// tensor_load_to_lds per K-step, double-buffered, TENSORcnt-synchronized).
// TDM zero-fills out-of-tensor reads, so K tails (K=144) and partial M/N
// tiles need no guards; the epilogue masks out-of-range stores.
__global__ __launch_bounds__(256)
void k_gemm(const bf16_t* __restrict__ A, long lda, long sA,
            const bf16_t* __restrict__ Bw, long ldb, long sB,
            float* __restrict__ Cf, bf16_t* __restrict__ Cb, long ldc, long sC,
            const float* __restrict__ bias,
            const float* __restrict__ Res, long ldr, long sR,
            int M, int N, int K, int act)
{
  const int bz = blockIdx.z;
  A  += (long)bz * sA;
  Bw += (long)bz * sB;
  if (Cf)  Cf  += (long)bz * sC;
  if (Cb)  Cb  += (long)bz * sC;
  if (Res) Res += (long)bz * sR;

  const int m0 = blockIdx.y * 64;
  const int n0 = blockIdx.x * 64;

  __shared__ bf16_t As[2][64][32];
  __shared__ bf16_t Bs[2][64][32];

  const int tid  = threadIdx.x;
  const int lane = tid & 31;
  const int wm   = (tid >> 5) >> 1;   // 0..3  (16-row strip)
  const int wn   = (tid >> 5) & 1;    // 0..1  (32-col strip, 2 frags)
  const int hf   = lane >> 4;
  const int r16  = lane & 15;

  const bf16_t* Abase = A  + (long)m0 * lda;
  const bf16_t* Bbase = Bw + (long)n0 * ldb;
  const int mrem = M - m0;
  const int nrem = N - n0;

  v8f acc[2];
#pragma unroll
  for (int i = 0; i < 8; ++i) { acc[0][i] = 0.0f; acc[1][i] = 0.0f; }

  const int nk = (K + 31) >> 5;        // number of 32-wide K chunks

  if (tid < 32) {                      // wave 0 drives the TDM
    tdm_tile_load((unsigned)(__SIZE_TYPE__)&As[0][0][0], Abase, mrem, K, lda);
    tdm_tile_load((unsigned)(__SIZE_TYPE__)&Bs[0][0][0], Bbase, nrem, K, ldb);
  }

  for (int t = 0; t < nk; ++t) {
    const int cur = t & 1;
    if (tid < 32) {
      if (t + 1 < nk) {                // prefetch next pair into other buffer
        const int kn = (t + 1) * 32;
        tdm_tile_load((unsigned)(__SIZE_TYPE__)&As[cur ^ 1][0][0],
                      Abase + kn, mrem, K - kn, lda);
        tdm_tile_load((unsigned)(__SIZE_TYPE__)&Bs[cur ^ 1][0][0],
                      Bbase + kn, nrem, K - kn, ldb);
        __builtin_amdgcn_s_wait_tensorcnt((short)2);   // current pair landed
      } else {
        __builtin_amdgcn_s_wait_tensorcnt((short)0);
      }
    }
    __syncthreads();

    v16bf a, b0, b1;
    {
      // A frag: lane<16 K{0..15}, lane>=16 K{16..31}
      const bf16_t* ap = &As[cur][16 * wm + r16][hf * 16];
#pragma unroll
      for (int j = 0; j < 16; ++j) { a[j] = ap[j]; }
      // B frags: lane<16 K{0..15}, lane>=16 K{16..31}; lane%16 = N col
      const bf16_t* q0 = &Bs[cur][32 * wn + r16][hf * 16];
#pragma unroll
      for (int j = 0; j < 16; ++j) b0[j] = q0[j];
      const bf16_t* q1 = &Bs[cur][32 * wn + 16 + r16][hf * 16];
#pragma unroll
      for (int j = 0; j < 16; ++j) b1[j] = q1[j];
    }
    acc[0] = __builtin_amdgcn_wmma_f32_16x16x32_bf16(false, a, false, b0,
                                                     (short)0, acc[0], false, false);
    acc[1] = __builtin_amdgcn_wmma_f32_16x16x32_bf16(false, a, false, b1,
                                                     (short)0, acc[1], false, false);
    __syncthreads();
  }

  // ---- epilogue (C layout: vgpr i -> M = i + 8*(lane>=16), N = lane%16) ----
  const int c16 = lane & 15;
#pragma unroll
  for (int f = 0; f < 2; ++f) {
    const int c = n0 + 32 * wn + 16 * f + c16;
    if (c >= N) continue;
    const float bv = bias ? bias[c] : 0.0f;
#pragma unroll
    for (int i = 0; i < 8; ++i) {
      const int r = m0 + 16 * wm + 8 * hf + i;
      if (r >= M) continue;
      float v = acc[f][i] + bv;
      if (Res) v += Res[(long)r * ldr + c];
      if (act == 1)      v = fmaxf(v, 0.0f);
      else if (act == 2) v = sigm(v);
      const long ci = (long)r * ldc + c;
      if (Cf) Cf[ci] = v;
      if (Cb) Cb[ci] = f2b(v);
    }
  }
}

// ------------------------------ LayerNorm ----------------------------------
// one block (256 threads) per row; E == 256 everywhere here.
__global__ __launch_bounds__(256)
void k_ln(const float* __restrict__ X, const float* __restrict__ g,
          const float* __restrict__ b, float* __restrict__ Yf,
          bf16_t* __restrict__ Yb)
{
  const int r = blockIdx.x, t = threadIdx.x;
  float v = X[(long)r * 256 + t];
  __shared__ float sd[256];
  sd[t] = v; __syncthreads();
  for (int s2 = 128; s2 > 0; s2 >>= 1) { if (t < s2) sd[t] += sd[t + s2]; __syncthreads(); }
  const float mean = sd[0] * (1.0f / 256.0f);
  __syncthreads();
  const float d = v - mean;
  sd[t] = d * d; __syncthreads();
  for (int s2 = 128; s2 > 0; s2 >>= 1) { if (t < s2) sd[t] += sd[t + s2]; __syncthreads(); }
  const float var = sd[0] * (1.0f / 256.0f);
  const float y = d * rsqrtf(var + 1e-5f) * g[t] + b[t];
  Yf[(long)r * 256 + t] = y;
  Yb[(long)r * 256 + t] = f2b(y);
}

// ----------------------------- softmax (T=256) -----------------------------
__global__ __launch_bounds__(256)
void k_softmax(const float* __restrict__ Att, bf16_t* __restrict__ Out)
{
  const int row = blockIdx.x, t = threadIdx.x;
  const float v = Att[(long)row * 256 + t];
  __shared__ float sd[256];
  sd[t] = v; __syncthreads();
  for (int s2 = 128; s2 > 0; s2 >>= 1) { if (t < s2) sd[t] = fmaxf(sd[t], sd[t + s2]); __syncthreads(); }
  const float mx = sd[0];
  __syncthreads();
  const float e = __expf(v - mx);
  sd[t] = e; __syncthreads();
  for (int s2 = 128; s2 > 0; s2 >>= 1) { if (t < s2) sd[t] += sd[t + s2]; __syncthreads(); }
  Out[(long)row * 256 + t] = f2b(e / sd[0]);
}

// -------------------------- Sinkhorn (per-batch) ---------------------------
__global__ __launch_bounds__(256)
void k_sinkhorn(const float* __restrict__ zmat, const float* __restrict__ gum,
                bf16_t* __restrict__ perm)
{
  const int b = blockIdx.x, t = threadIdx.x;
  __shared__ float la[64][65];
  for (int e = t; e < 4096; e += 256) {
    const int i = e >> 6, j = e & 63;
    la[i][j] = (zmat[(long)b * 4096 + e] + gum[(long)b * 4096 + e]) * 10.0f; // /TEMP
  }
  __syncthreads();
  for (int it = 0; it < 20; ++it) {
    if (t < 64) {                    // logsumexp over last axis (row t)
      float m = -1e30f;
      for (int j = 0; j < 64; ++j) m = fmaxf(m, la[t][j]);
      float s = 0.0f;
      for (int j = 0; j < 64; ++j) s += __expf(la[t][j] - m);
      const float l = m + __logf(s);
      for (int j = 0; j < 64; ++j) la[t][j] -= l;
    }
    __syncthreads();
    if (t < 64) {                    // logsumexp over axis 1 (col t)
      float m = -1e30f;
      for (int i = 0; i < 64; ++i) m = fmaxf(m, la[i][t]);
      float s = 0.0f;
      for (int i = 0; i < 64; ++i) s += __expf(la[i][t] - m);
      const float l = m + __logf(s);
      for (int i = 0; i < 64; ++i) la[i][t] -= l;
    }
    __syncthreads();
  }
  for (int e = t; e < 4096; e += 256) {
    const int i = e >> 6, j = e & 63;
    perm[(long)b * 4096 + e] = f2b(__expf(la[i][j]));
  }
}

// --------------------- g_in identity columns (192..255) --------------------
__global__ void k_gin_eye(float* __restrict__ Gf, bf16_t* __restrict__ Gb)
{
  const int idx = blockIdx.x * blockDim.x + threadIdx.x;   // 256 rows * 64 cols
  const int row = idx >> 6, j = idx & 63;
  const float v = (j == (row & 63)) ? 1.0f : 0.0f;
  Gf[(long)row * 256 + 192 + j] = v;
  Gb[(long)row * 256 + 192 + j] = f2b(v);
}

// ------------------- f_in assembly (token rows s*65 + n) -------------------
// Fe: [256,176] rows b*64+t ; Yhat: [256,16] rows b*64+t
__global__ void k_fin(const float* __restrict__ Fe, const float* __restrict__ Yhat,
                      float* __restrict__ Ff, bf16_t* __restrict__ Fb)
{
  const long total = 16640L * 256;
  long i  = (long)blockIdx.x * blockDim.x + threadIdx.x;
  long st = (long)gridDim.x * blockDim.x;
  for (; i < total; i += st) {
    const int c  = (int)(i & 255);
    const long tr = i >> 8;
    const int s = (int)(tr / 65);       // = b*64 + l
    const int n = (int)(tr % 65);
    const int b = s >> 6;
    float v;
    if (n < 64) {                       // f_seq[b, n]
      const int rb = (b << 6) + n;
      if (c < 176)       v = Fe[(long)rb * 176 + c];
      else if (c < 192)  v = Yhat[(long)rb * 16 + (c - 176)];
      else               v = (c - 192 == n) ? 1.0f : 0.0f;
    } else {                            // tok[b, l] : cols >= 65 forced to 1
      if (c >= 65) v = 1.0f;
      else         v = Fe[(long)s * 176 + c];
    }
    Ff[i] = v;
    Fb[i] = f2b(v);
  }
}

// -------- split qkv (f-enc) into per-head Q,K bf16 and V^T bf16 ------------
// qkv: [S*N, 768] rows s*N+n ; Q,K: [(n*8+h), S, 32] ; VT: [(n*8+h), 32, S]
__global__ void k_qkv_split(const float* __restrict__ qkv,
                            bf16_t* __restrict__ Q, bf16_t* __restrict__ Kh,
                            bf16_t* __restrict__ VT, int S, int N, float qscale)
{
  const long total = (long)N * 8 * S * 32;
  long i  = (long)blockIdx.x * blockDim.x + threadIdx.x;
  long st = (long)gridDim.x * blockDim.x;
  for (; i < total; i += st) {
    const int d = (int)(i & 31);
    long t = i >> 5;
    const int s = (int)(t % S); t /= S;
    const int h = (int)(t & 7);
    const int n = (int)(t >> 3);
    const long row = (long)s * N + n;
    const float* base = qkv + row * 768;
    const int col = h * 32 + d;
    const long bh = (long)n * 8 + h;
    Q [(bh * S + s) * 32 + d]        = f2b(base[col] * qscale);
    Kh[(bh * S + s) * 32 + d]        = f2b(base[256 + col]);
    VT[bh * 32 * (long)S + (long)d * S + s] = f2b(base[512 + col]);
  }
}

// ---------------- merge heads: O [(n*8+h),S,32] -> Mg [S*N,256] ------------
__global__ void k_merge_heads(const float* __restrict__ O, bf16_t* __restrict__ Mg,
                              int S, int N)
{
  const long total = (long)N * 8 * S * 32;
  long i  = (long)blockIdx.x * blockDim.x + threadIdx.x;
  long st = (long)gridDim.x * blockDim.x;
  for (; i < total; i += st) {
    const int d = (int)(i & 31);
    long t = i >> 5;
    const int s = (int)(t % S); t /= S;
    const int h = (int)(t & 7);
    const int n = (int)(t >> 3);
    Mg[((long)s * N + n) * 256 + h * 32 + d] =
        f2b(O[(((long)n * 8 + h) * S + s) * 32 + d]);
  }
}

// --------------- g-encoder attention (S=4, N=64, 8 heads) VALU -------------
// qkv: [256,768] rows s*64+n ; Mg: [256,256] rows s*64+n
__global__ void k_attn_g(const float* __restrict__ qkv, bf16_t* __restrict__ Mg)
{
  const int idx = blockIdx.x * blockDim.x + threadIdx.x;   // (n,h,s)
  if (idx >= 2048) return;
  const int s = idx & 3;
  const int h = (idx >> 2) & 7;
  const int n = idx >> 5;
  const float scale = 0.17677669529663687f;                // 1/sqrt(32)
  float att[4];
  const float* qp = qkv + ((long)(s * 64 + n)) * 768 + h * 32;
  for (int t2 = 0; t2 < 4; ++t2) {
    const float* kp = qkv + ((long)(t2 * 64 + n)) * 768 + 256 + h * 32;
    float d = 0.0f;
    for (int d2 = 0; d2 < 32; ++d2) d += qp[d2] * kp[d2];
    att[t2] = d * scale;
  }
  float mx = fmaxf(fmaxf(att[0], att[1]), fmaxf(att[2], att[3]));
  float se = 0.0f;
  for (int t2 = 0; t2 < 4; ++t2) { att[t2] = __expf(att[t2] - mx); se += att[t2]; }
  const float inv = 1.0f / se;
  for (int t2 = 0; t2 < 4; ++t2) att[t2] *= inv;
  for (int d2 = 0; d2 < 32; ++d2) {
    float o = 0.0f;
    for (int t2 = 0; t2 < 4; ++t2)
      o += att[t2] * qkv[((long)(t2 * 64 + n)) * 768 + 512 + h * 32 + d2];
    Mg[((long)(s * 64 + n)) * 256 + h * 32 + d2] = f2b(o);
  }
}

// ------------- rho reduce + gru_in assembly (uses ORIGINAL x) --------------
// Phi: [256,16] rows b*64+l ; X: [256,128] f32 ; Grin: [256,144] bf16
__global__ __launch_bounds__(256)
void k_rho_gruin(const float* __restrict__ Phi, const float* __restrict__ X,
                 bf16_t* __restrict__ Grin)
{
  const int b = blockIdx.x, t = threadIdx.x;
  __shared__ float rho[16];
  if (t < 16) {
    float s = 0.0f;
    for (int l = 0; l < 64; ++l) s += Phi[((long)b * 64 + l) * 16 + t];
    rho[t] = s;
  }
  __syncthreads();
  for (int e = t; e < 64 * 144; e += 256) {
    const int l = e / 144, c = e % 144;
    const long r = (long)b * 64 + l;
    float v;
    if (c < 128) v = X[r * 128 + c];
    else {
      // replicate reference's flat repeat: rho_rep[b,l,j2] = rho[b, (l*16+j2)//64]
      const int j = ((l * 16 + (c - 128)) >> 6);
      v = rho[j];
    }
    Grin[r * 144 + c] = f2b(v);
  }
}

// ----------------------------- GRU recurrence ------------------------------
// Xp: [256,768] rows b*64+t (precomputed x@w_ih^T + b_ih); one block per batch.
__global__ __launch_bounds__(256)
void k_gru(const float* __restrict__ Xp, const float* __restrict__ Whh,
           const float* __restrict__ Bhh, bf16_t* __restrict__ Hb)
{
  const int b = blockIdx.x, t = threadIdx.x;
  __shared__ float h[256];
  __shared__ float gh[768];
  h[t] = 0.0f;
  __syncthreads();
  for (int step = 0; step < 64; ++step) {
    for (int jj = 0; jj < 3; ++jj) {
      const int j = t + jj * 256;
      const float* wr = Whh + (long)j * 256;
      float s = Bhh[j];
      for (int k2 = 0; k2 < 256; ++k2) s += wr[k2] * h[k2];
      gh[j] = s;
    }
    __syncthreads();
    const long row = (long)b * 64 + step;
    const float* gi = Xp + row * 768;
    const float r  = sigm(gi[t] + gh[t]);
    const float z  = sigm(gi[256 + t] + gh[256 + t]);
    const float nn = tanhf(gi[512 + t] + r * gh[512 + t]);
    const float hn = (1.0f - z) * nn + z * h[t];
    __syncthreads();
    h[t] = hn;
    Hb[row * 256 + t] = f2b(hn);
    __syncthreads();
  }
}

// ----------------------------- host GEMM launch ----------------------------
static void gemm(hipStream_t st,
                 const bf16_t* A, long lda, long sA,
                 const bf16_t* Bw, long ldb, long sB,
                 float* Cf, bf16_t* Cb, long ldc, long sC,
                 const float* bias, const float* Res, long ldr, long sR,
                 int M, int N, int K, int act, int batch)
{
  dim3 grid((unsigned)((N + 63) / 64), (unsigned)((M + 63) / 64), (unsigned)batch);
  k_gemm<<<grid, 256, 0, st>>>(A, lda, sA, Bw, ldb, sB, Cf, Cb, ldc, sC,
                               bias, Res, ldr, sR, M, N, K, act);
}

static inline int gsblocks(long n) {
  long b = (n + 255) / 256;
  return (int)(b > 8192 ? 8192 : b);
}

// ---------------------------------------------------------------------------
extern "C" void kernel_launch(void* const* d_in, const int* in_sizes, int n_in,
                              void* d_out, int out_size, void* d_ws, size_t ws_size,
                              hipStream_t stream)
{
  (void)in_sizes; (void)n_in; (void)out_size; (void)ws_size;

  // ---------------- inputs (setup_inputs insertion order) ----------------
  const float* X   = (const float*)d_in[0];
  const float* GUM = (const float*)d_in[1];
  int pi = 2;
  auto nx = [&]() { return (const float*)d_in[pi++]; };
  const float* mat_w  = nx(); const float* mat_b  = nx();
  const float* gemb_w = nx(); const float* gemb_b = nx();
  const float* femb_w = nx(); const float* femb_b = nx();
  struct EncP {
    const float *in_w, *in_b, *out_w, *out_b, *ff1_w, *ff1_b, *ff2_w, *ff2_b,
                *ln1_g, *ln1_b, *ln2_g, *ln2_b;
  };
  EncP gl[2], fl[2];
  for (int l = 0; l < 2; ++l) {
    EncP& p = gl[l];
    p.in_w = nx(); p.in_b = nx(); p.out_w = nx(); p.out_b = nx();
    p.ff1_w = nx(); p.ff1_b = nx(); p.ff2_w = nx(); p.ff2_b = nx();
    p.ln1_g = nx(); p.ln1_b = nx(); p.ln2_g = nx(); p.ln2_b = nx();
  }
  for (int l = 0; l < 2; ++l) {
    EncP& p = fl[l];
    p.in_w = nx(); p.in_b = nx(); p.out_w = nx(); p.out_b = nx();
    p.ff1_w = nx(); p.ff1_b = nx(); p.ff2_w = nx(); p.ff2_b = nx();
    p.ln1_g = nx(); p.ln1_b = nx(); p.ln2_g = nx(); p.ln2_b = nx();
  }
  const float* gp_w  = nx(); const float* gp_b  = nx();
  const float* phi_w = nx(); const float* phi_b = nx();
  struct GruP { const float *w_ih, *w_hh, *b_ih, *b_hh; } gr[2];
  for (int l = 0; l < 2; ++l) { gr[l].w_ih = nx(); gr[l].w_hh = nx(); gr[l].b_ih = nx(); gr[l].b_hh = nx(); }
  const float* fc_w = nx(); const float* fc_b = nx();

  // ---------------- workspace bump allocator ----------------
  char* wsb = (char*)d_ws;
  size_t off = 0;
  auto alloc = [&](size_t bytes) -> void* {
    void* p = wsb + off;
    off += bytes;
    off = (off + 255) & ~(size_t)255;
    return p;
  };
  auto cvtw = [&](const float* src, long n) -> bf16_t* {
    bf16_t* d = (bf16_t*)alloc((size_t)n * 2);
    k_cvt<<<gsblocks(n), 256, 0, stream>>>(src, d, n);
    return d;
  };

  // ---------------- weights -> bf16 (once per call) ----------------
  bf16_t* mat_wb  = cvtw(mat_w, 128 * 128);
  bf16_t* gemb_wb = cvtw(gemb_w, 192 * 128);
  bf16_t* femb_wb = cvtw(femb_w, 176 * 128);
  struct EncWB { bf16_t *in_w, *out_w, *ff1_w, *ff2_w; } glb[2], flb[2];
  for (int l = 0; l < 2; ++l) {
    glb[l].in_w  = cvtw(gl[l].in_w, 768L * 256);
    glb[l].out_w = cvtw(gl[l].out_w, 256L * 256);
    glb[l].ff1_w = cvtw(gl[l].ff1_w, 2048L * 256);
    glb[l].ff2_w = cvtw(gl[l].ff2_w, 256L * 2048);
  }
  for (int l = 0; l < 2; ++l) {
    flb[l].in_w  = cvtw(fl[l].in_w, 768L * 256);
    flb[l].out_w = cvtw(fl[l].out_w, 256L * 256);
    flb[l].ff1_w = cvtw(fl[l].ff1_w, 2048L * 256);
    flb[l].ff2_w = cvtw(fl[l].ff2_w, 256L * 2048);
  }
  bf16_t* gp_wb  = cvtw(gp_w, 16 * 256);
  bf16_t* phi_wb = cvtw(phi_w, 16 * 256);
  bf16_t* wih0b  = cvtw(gr[0].w_ih, 768L * 144);
  bf16_t* wih1b  = cvtw(gr[1].w_ih, 768L * 256);
  bf16_t* fc_wb  = cvtw(fc_w, 16 * 256);

  // ---------------- activation buffers ----------------
  const long RF = 16640;                         // f-enc token rows (256*65)
  bf16_t* xbf   = (bf16_t*)alloc(32768L * 2);
  bf16_t* xTbf  = (bf16_t*)alloc(32768L * 2);
  float*  zf    = (float*)alloc(32768L * 4);
  bf16_t* zbf   = (bf16_t*)alloc(32768L * 2);
  float*  zmat  = (float*)alloc(16384L * 4);
  bf16_t* permb = (bf16_t*)alloc(16384L * 2);
  float*  xwb   = (float*)alloc(32768L * 4);
  bf16_t* xwbb  = (bf16_t*)alloc(32768L * 2);

  float*  act   = (float*)alloc(RF * 256 * 4);
  bf16_t* actb  = (bf16_t*)alloc(RF * 256 * 2);
  float*  lnin  = (float*)alloc(RF * 256 * 4);
  float*  qkvf  = (float*)alloc(RF * 768 * 4);
  bf16_t* qh    = (bf16_t*)alloc(520L * 256 * 32 * 2);
  bf16_t* kh    = (bf16_t*)alloc(520L * 256 * 32 * 2);
  bf16_t* vT    = (bf16_t*)alloc(520L * 32 * 256 * 2);
  float*  attf  = (float*)alloc(104L * 256 * 256 * 4);
  bf16_t* attb  = (bf16_t*)alloc(104L * 256 * 256 * 2);
  float*  obuf  = (float*)alloc(520L * 256 * 32 * 4);
  bf16_t* mrgb  = (bf16_t*)alloc(RF * 256 * 2);
  bf16_t* hbf   = (bf16_t*)alloc(4160L * 2048 * 2);
  float*  fe    = (float*)alloc(256L * 176 * 4);
  float*  yhat1 = (float*)alloc(256L * 16 * 4);
  float*  phib  = (float*)alloc(256L * 16 * 4);
  bf16_t* grinb = (bf16_t*)alloc(256L * 144 * 2);
  float*  xp    = (float*)alloc(256L * 768 * 4);
  bf16_t* hb1   = (bf16_t*)alloc(256L * 256 * 2);
  bf16_t* hb2   = (bf16_t*)alloc(256L * 256 * 2);

  float* OUT = (float*)d_out;

  // ---------------- preamble: z, perm, x_wb ----------------
  k_cvt<<<128, 256, 0, stream>>>(X, xbf, 32768);
  k_transpose_bf<<<128, 256, 0, stream>>>(X, xTbf, 64, 128, 32768);

  // z = x @ mat_w^T + mat_b
  gemm(stream, xbf, 128, 0, mat_wb, 128, 0, zf, zbf, 128, 0,
       mat_b, nullptr, 0, 0, 256, 128, 128, 0, 1);
  // zmat[b] = z_b @ z_b^T
  gemm(stream, zbf, 128, 64 * 128, zbf, 128, 64 * 128, zmat, nullptr, 64, 64 * 64,
       nullptr, nullptr, 0, 0, 64, 64, 128, 0, 4);
  k_sinkhorn<<<4, 256, 0, stream>>>(zmat, GUM, permb);
  // x_wb[b] = perm_b @ x_b   (W = x_b^T, [128,64])
  gemm(stream, permb, 64, 64 * 64, xTbf, 64, 128 * 64, xwb, xwbb, 128, 64 * 128,
       nullptr, nullptr, 0, 0, 64, 128, 64, 0, 4);

  // ---------------- encoder layer ----------------
  auto encLayer = [&](const EncP& P, const EncWB& W, bool big) {
    const int R = big ? (int)RF : 256;
    // qkv
    gemm(stream, actb, 256, 0, W.in_w, 256, 0, qkvf, nullptr, 768, 0,
         P.in_b, nullptr, 0, 0, R, 768, 256, 0, 1);
    if (!big) {
      k_attn_g<<<8, 256, 0, stream>>>(qkvf, mrgb);
    } else {
      const long tot = 520L * 256 * 32;
      k_qkv_split<<<gsblocks(tot), 256, 0, stream>>>(qkvf, qh, kh, vT, 256, 65,
                                                     0.17677669529663687f);
      for (int c = 0; c < 5; ++c) {                 // 5 chunks of 13 positions
        const long hb = 104L * c;
        // scores: [104, 256, 256]
        gemm(stream, qh + hb * 8192, 32, 8192, kh + hb * 8192, 32, 8192,
             attf, nullptr, 256, 65536, nullptr, nullptr, 0, 0,
             256, 256, 32, 0, 104);
        k_softmax<<<104 * 256, 256, 0, stream>>>(attf, attb);
        // context: [104, 256, 32]
        gemm(stream, attb, 256, 65536, vT + hb * 8192, 256, 8192,
             obuf + hb * 8192, nullptr, 32, 8192, nullptr, nullptr, 0, 0,
             256, 32, 256, 0, 104);
      }
      k_merge_heads<<<gsblocks(tot), 256, 0, stream>>>(obuf, mrgb, 256, 65);
    }
    // out-projection + residual
    gemm(stream, mrgb, 256, 0, W.out_w, 256, 0, lnin, nullptr, 256, 0,
         P.out_b, act, 256, 0, R, 256, 256, 0, 1);
    k_ln<<<R, 256, 0, stream>>>(lnin, P.ln1_g, P.ln1_b, act, actb);
    // FF (row-chunked so the 2048-wide hidden stays bf16 and bounded)
    const int rc = big ? 4160 : R;
    for (int o2 = 0; o2 < R; o2 += rc) {
      gemm(stream, actb + (long)o2 * 256, 256, 0, W.ff1_w, 256, 0,
           nullptr, hbf, 2048, 0, P.ff1_b, nullptr, 0, 0, rc, 2048, 256, 1, 1);
      gemm(stream, hbf, 2048, 0, W.ff2_w, 2048, 0,
           lnin + (long)o2 * 256, nullptr, 256, 0, P.ff2_b,
           act + (long)o2 * 256, 256, 0, rc, 256, 2048, 0, 1);
    }
    k_ln<<<R, 256, 0, stream>>>(lnin, P.ln2_g, P.ln2_b, act, actb);
  };

  // ---------------- one branch ----------------
  auto runBranch = [&](const bf16_t* xinb, float* yDst, float* yhatDst) {
    // g_in = [xin@gemb^T + b | eye]
    gemm(stream, xinb, 128, 0, gemb_wb, 128, 0, act, actb, 256, 0,
         gemb_b, nullptr, 0, 0, 256, 192, 128, 0, 1);
    k_gin_eye<<<64, 256, 0, stream>>>(act, actb);
    encLayer(gl[0], glb[0], false);
    encLayer(gl[1], glb[1], false);
    // yhat = sigmoid(g_out @ gp^T + b)
    gemm(stream, actb, 256, 0, gp_wb, 256, 0, yhatDst, nullptr, 16, 0,
         gp_b, nullptr, 0, 0, 256, 16, 256, 2, 1);
    // fe = xin @ femb^T + b
    gemm(stream, xinb, 128, 0, femb_wb, 128, 0, fe, nullptr, 176, 0,
         femb_b, nullptr, 0, 0, 256, 176, 128, 0, 1);
    k_fin<<<gsblocks(16640L * 256), 256, 0, stream>>>(fe, yhatDst, act, actb);
    encLayer(fl[0], flb[0], true);
    encLayer(fl[1], flb[1], true);
    // phi = relu(f_out[:,64,:] @ phi^T + b)  (strided A: token rows n==64)
    gemm(stream, actb + 64L * 256, 65L * 256, 0, phi_wb, 256, 0,
         phib, nullptr, 16, 0, phi_b, nullptr, 0, 0, 256, 16, 256, 1, 1);
    // rho + gru_in (uses ORIGINAL x in both branches, per reference)
    k_rho_gruin<<<4, 256, 0, stream>>>(phib, X, grinb);
    // GRU layer 0
    gemm(stream, grinb, 144, 0, wih0b, 144, 0, xp, nullptr, 768, 0,
         gr[0].b_ih, nullptr, 0, 0, 256, 768, 144, 0, 1);
    k_gru<<<4, 256, 0, stream>>>(xp, gr[0].w_hh, gr[0].b_hh, hb1);
    // GRU layer 1
    gemm(stream, hb1, 256, 0, wih1b, 256, 0, xp, nullptr, 768, 0,
         gr[1].b_ih, nullptr, 0, 0, 256, 768, 256, 0, 1);
    k_gru<<<4, 256, 0, stream>>>(xp, gr[1].w_hh, gr[1].b_hh, hb2);
    // y = g @ fc^T + b
    gemm(stream, hb2, 256, 0, fc_wb, 256, 0, yDst, nullptr, 16, 0,
         fc_b, nullptr, 0, 0, 256, 16, 256, 0, 1);
  };

  // y -> OUT[0..4095], y_wb -> OUT[4096..8191], yhat -> OUT[8192..12287]
  runBranch(xbf,  OUT + 0,    OUT + 8192);
  runBranch(xwbb, OUT + 4096, yhat1);
}
